// LSTM_50989851738665
// MI455X (gfx1250) — matrix operand order repaired and checked by reading, not compiled, not run on hardware
//
#include <hip/hip_runtime.h>
#include <hip/hip_bf16.h>

#define T_STEPS 8192
#define E_DIM   1024
#define H_DIM   1024
#define NWG_REC 32
#define ROWS_PER_WG (H_DIM / NWG_REC)   // 32 rows per workgroup

typedef __attribute__((ext_vector_type(16))) __bf16       v16bf;
typedef __attribute__((ext_vector_type(8)))  float        v8f;
typedef __attribute__((ext_vector_type(4)))  unsigned int uint4v;

// ---------------------------------------------------------------------------
// CDNA5 async global->LDS copy (ASYNCcnt-tracked, no VGPR round-trip).
// LDS destination address is the low 32 bits of the generic pointer
// (ISA: LDS aperture -> addr[31:0]).
// ---------------------------------------------------------------------------
__device__ inline void async_copy_b128(void* lds_dst, const void* gsrc) {
  unsigned lds_off = (unsigned)(unsigned long long)lds_dst;
  asm volatile("global_load_async_to_lds_b128 %0, %1, off"
               :: "v"(lds_off), "v"(gsrc)
               : "memory");
}
__device__ inline void wait_async0() {
  asm volatile("s_wait_asynccnt 0x0" ::: "memory");
}

// ---------------------------------------------------------------------------
// f32 -> bf16 (RNE) conversion, grid-stride
// ---------------------------------------------------------------------------
__global__ __launch_bounds__(256) void cvt_f32_bf16(const float* __restrict__ in,
                                                    unsigned short* __restrict__ out,
                                                    int n) {
  for (int i = blockIdx.x * blockDim.x + threadIdx.x; i < n;
       i += gridDim.x * blockDim.x) {
    unsigned u = __float_as_uint(in[i]);
    out[i] = (unsigned short)((u + 0x7FFFu + ((u >> 16) & 1u)) >> 16);
  }
}

// ---------------------------------------------------------------------------
// Zero-fill (c_out), grid-stride
// ---------------------------------------------------------------------------
__global__ __launch_bounds__(256) void zero_f32(float* __restrict__ p, int n) {
  for (int i = blockIdx.x * blockDim.x + threadIdx.x; i < n;
       i += gridDim.x * blockDim.x)
    p[i] = 0.0f;
}

// ---------------------------------------------------------------------------
// Init: hbuf[0] = h0, barrier counter = 0
// ---------------------------------------------------------------------------
__global__ __launch_bounds__(1024) void init_rec(const float* __restrict__ h0,
                                                 float* __restrict__ hbuf,
                                                 unsigned int* __restrict__ cnt) {
  int t = threadIdx.x;
  if (t < H_DIM) hbuf[t] = h0[t];
  if (t == 0) *cnt = 0u;
}

// ---------------------------------------------------------------------------
// Input-projection GEMMs:  pre[g][t][h] = sum_e X[t,e] * W_g[h,e]
// bf16 WMMA, f32 accumulate. Block = 128 threads (4 waves).
// Each wave computes a 64(M) x 16(N) strip: 4 A frags + 1 B frag + 4 WMMA / K.
// A (16x32 bf16): lane<16 -> row M=lane, K runs {8*hl .. 8*hl+7, 16+8*hl ..}
// B (32x16 bf16): lane<16 -> col N=lane, K runs {8*hl .. 8*hl+7, 16+8*hl ..}
// C (16x16 f32):  lane<16 -> N=lane, VGPR r = M=r; lane>=16 -> M=r+8
// ---------------------------------------------------------------------------
__global__ __launch_bounds__(128) void wmma_gate_gemm(
    const unsigned short* __restrict__ Xbf,   // [T,E] bf16
    const unsigned short* __restrict__ Wbf,   // [4][H,E] bf16
    float* __restrict__ pre)                  // [4][T,H] f32
{
  const int lane = threadIdx.x & 31;
  const int wave = threadIdx.x >> 5;
  const int g    = blockIdx.z;
  const int m0   = blockIdx.x * 64;   // T block
  const int n0   = blockIdx.y * 64;   // H block

  const unsigned short* Wg   = Wbf + (size_t)g * H_DIM * E_DIM;
  float*                outg = pre + (size_t)g * T_STEPS * H_DIM;

  const int hl = lane >> 4;     // half-wave select
  const int mr = lane & 15;     // A-row / B-col within tile

  v8f acc[4];
  #pragma unroll
  for (int i = 0; i < 4; ++i)
    #pragma unroll
    for (int j = 0; j < 8; ++j) acc[i][j] = 0.0f;

  // Per-lane base pointers matching the WMMA VGPR layouts
  const unsigned short* Brow = Wg + (size_t)(n0 + wave * 16 + mr) * E_DIM + 8 * hl;

  for (int k0 = 0; k0 < E_DIM; k0 += 32) {
    union { uint4v u[2]; v16bf v; } bu;
    bu.u[0] = *(const uint4v*)(Brow + k0);
    bu.u[1] = *(const uint4v*)(Brow + k0 + 16);
    __builtin_prefetch(Brow + k0 + 32, 0, 2);

    #pragma unroll
    for (int i = 0; i < 4; ++i) {
      const unsigned short* Arow =
          Xbf + (size_t)(m0 + i * 16 + mr) * E_DIM + k0 + 8 * hl;
      union { uint4v u[2]; v16bf v; } au;
      au.u[0] = *(const uint4v*)(Arow);
      au.u[1] = *(const uint4v*)(Arow + 16);
      acc[i] = __builtin_amdgcn_wmma_f32_16x16x32_bf16(
          false, au.v, false, bu.v, (short)0, acc[i], false, false);
    }
  }

  const int colg = n0 + wave * 16 + mr;
  #pragma unroll
  for (int i = 0; i < 4; ++i) {
    const int mbase = m0 + i * 16 + hl * 8;
    #pragma unroll
    for (int r = 0; r < 8; ++r)
      outg[(size_t)(mbase + r) * H_DIM + colg] = acc[i][r];
  }
}

// ---------------------------------------------------------------------------
// Persistent recurrence kernel. 32 WGs x 256 threads; each WG owns 32 rows.
// LDS: U slice (4 gates x 32 rows x 1024 bf16 = 256 KB) + h (4 KB f32)
//      + 128 pre-activations + 32 cell states.   (~261 KB of the 320 KB WGP LDS)
// U slice and per-step h broadcast staged with GLOBAL_LOAD_ASYNC_TO_LDS_B128
// (ASYNCcnt), dot products are wave-cooperative with conflict-free LDS strides,
// steps separated by a device-scope atomic barrier on a monotonic counter.
// ---------------------------------------------------------------------------
__global__ __launch_bounds__(256) void lstm_recurrence(
    const unsigned short* __restrict__ Ubf,   // [4][H,H] bf16
    const float* __restrict__ pre,            // [4][T,H] f32
    const float* __restrict__ c0,             // [H]
    float* __restrict__ hbuf,                 // [2][H]
    unsigned int* __restrict__ barrier_cnt,
    float* __restrict__ h_out)                // [T,H]
{
  extern __shared__ char smem[];
  unsigned short* U_lds   = (unsigned short*)smem;                 // [128][1024]
  float*          h_lds   = (float*)(smem + 128 * 1024 * 2);       // [1024]
  float*          pre_lds = h_lds + H_DIM;                         // [128]
  float*          c_lds   = pre_lds + 128;                         // [32]

  const int tid  = threadIdx.x;
  const int wg   = blockIdx.x;
  const int r0   = wg * ROWS_PER_WG;
  const int lane = tid & 31;
  const int wave = tid >> 5;

  // Stage this WG's U slice into LDS via async b128 copies (pair p = g*32+r)
  for (int q = tid; q < 128 * 128; q += 256) {
    const int p  = q >> 7;        // 0..127  (g*32+r)
    const int i  = q & 127;       // b128 index within row
    const int gg = p >> 5, rr = p & 31;
    const unsigned short* src =
        Ubf + ((size_t)gg * H_DIM + (r0 + rr)) * H_DIM + i * 8;
    async_copy_b128(U_lds + (size_t)p * H_DIM + i * 8, src);
  }
  if (tid < ROWS_PER_WG) c_lds[tid] = c0[r0 + tid];
  wait_async0();
  __syncthreads();

  for (int t = 0; t < T_STEPS; ++t) {
    const float* hsrc = hbuf + (t & 1) * H_DIM;
    float*       hdst = hbuf + ((t + 1) & 1) * H_DIM;

    // Broadcast h_{t-1} into LDS (256 threads x b128 = 1024 floats), async
    async_copy_b128(h_lds + tid * 4, hsrc + tid * 4);
    wait_async0();
    __syncthreads();

    // Wave w computes pairs p = w*16 .. w*16+15 (one dot product each)
    #pragma unroll 1
    for (int pp = 0; pp < 16; ++pp) {
      const int p = wave * 16 + pp;
      const unsigned short* urow = U_lds + (size_t)p * H_DIM;
      float acc = 0.0f;
      #pragma unroll
      for (int j = 0; j < 8; ++j) {
        const int k = lane * 4 + j * 128;            // conflict-free strides
        uint2  u  = *(const uint2*)(urow + k);
        float4 hv = *(const float4*)(h_lds + k);
        acc = fmaf(__uint_as_float(u.x << 16),         hv.x, acc);
        acc = fmaf(__uint_as_float(u.x & 0xFFFF0000u), hv.y, acc);
        acc = fmaf(__uint_as_float(u.y << 16),         hv.z, acc);
        acc = fmaf(__uint_as_float(u.y & 0xFFFF0000u), hv.w, acc);
      }
      #pragma unroll
      for (int off = 16; off > 0; off >>= 1)
        acc += __shfl_xor(acc, off, 32);
      if (lane == 0) pre_lds[p] = acc;
    }
    __syncthreads();

    // Pointwise gates + state update for this WG's 32 rows
    if (tid < ROWS_PER_WG) {
      const int    row = r0 + tid;
      const size_t tb  = (size_t)t * H_DIM + row;
      const float fx = pre[tb];
      const float gx = pre[(size_t)1 * T_STEPS * H_DIM + tb];
      const float ix = pre[(size_t)2 * T_STEPS * H_DIM + tb];
      const float ox = pre[(size_t)3 * T_STEPS * H_DIM + tb];
      const float fv = 1.0f / (1.0f + __expf(-(pre_lds[tid]      + fx)));
      const float gv = tanhf(           pre_lds[32 + tid] + gx);
      const float iv = 1.0f / (1.0f + __expf(-(pre_lds[64 + tid] + ix)));
      const float ov = 1.0f / (1.0f + __expf(-(pre_lds[96 + tid] + ox)));
      const float cv = gv * iv + c_lds[tid] * fv;
      c_lds[tid] = cv;
      const float hv = ov * tanhf(cv);
      hdst[row]  = hv;
      h_out[tb]  = hv;
      __threadfence();
    }
    __syncthreads();

    // Device-wide step barrier (monotonic counter, re-zeroed per launch)
    if (tid == 0) {
      __hip_atomic_fetch_add(barrier_cnt, 1u, __ATOMIC_RELEASE,
                             __HIP_MEMORY_SCOPE_AGENT);
      const unsigned target = (unsigned)NWG_REC * (unsigned)(t + 1);
      while (__hip_atomic_load(barrier_cnt, __ATOMIC_ACQUIRE,
                               __HIP_MEMORY_SCOPE_AGENT) < target)
        __builtin_amdgcn_s_sleep(1);
    }
    __syncthreads();
  }
}

// ---------------------------------------------------------------------------
// Host launcher
// ---------------------------------------------------------------------------
extern "C" void kernel_launch(void* const* d_in, const int* in_sizes, int n_in,
                              void* d_out, int out_size, void* d_ws, size_t ws_size,
                              hipStream_t stream) {
  (void)in_sizes; (void)n_in; (void)out_size; (void)ws_size;

  const float* inputs = (const float*)d_in[0];
  const float* h0     = (const float*)d_in[1];
  const float* c0     = (const float*)d_in[2];
  const float* U[4] = {(const float*)d_in[3], (const float*)d_in[5],
                       (const float*)d_in[7], (const float*)d_in[9]};
  const float* W[4] = {(const float*)d_in[4], (const float*)d_in[6],
                       (const float*)d_in[8], (const float*)d_in[10]};

  float* h_out = (float*)d_out;                                   // [T,H]
  float* c_out = h_out + (size_t)T_STEPS * H_DIM;                 // [T,H] -> zeros

  // Workspace layout
  char* ws = (char*)d_ws;
  unsigned short* Xbf = (unsigned short*)ws;                            // T*E bf16
  unsigned short* Wbf = Xbf + (size_t)T_STEPS * E_DIM;                  // 4*H*E bf16
  unsigned short* Ubf = Wbf + (size_t)4 * H_DIM * E_DIM;                // 4*H*H bf16
  float*          pre = (float*)(Ubf + (size_t)4 * H_DIM * H_DIM);     // 4*T*H f32
  float*          hbuf = pre + (size_t)4 * T_STEPS * H_DIM;            // 2*H f32
  unsigned int*   cnt  = (unsigned int*)(hbuf + 2 * H_DIM);            // barrier

  // 1) f32 -> bf16 conversions
  cvt_f32_bf16<<<512, 256, 0, stream>>>(inputs, Xbf, T_STEPS * E_DIM);
  for (int g = 0; g < 4; ++g) {
    cvt_f32_bf16<<<256, 256, 0, stream>>>(W[g], Wbf + (size_t)g * H_DIM * E_DIM,
                                          H_DIM * E_DIM);
    cvt_f32_bf16<<<256, 256, 0, stream>>>(U[g], Ubf + (size_t)g * H_DIM * H_DIM,
                                          H_DIM * H_DIM);
  }

  // 2) Four input-projection GEMMs via bf16 WMMA (grid.z = gate)
  wmma_gate_gemm<<<dim3(T_STEPS / 64, H_DIM / 64, 4), 128, 0, stream>>>(Xbf, Wbf,
                                                                        pre);

  // 3) Init recurrence state (h ping-pong buffer + barrier counter) and c_out=0
  init_rec<<<1, 1024, 0, stream>>>(h0, hbuf, cnt);
  zero_f32<<<1024, 256, 0, stream>>>(c_out, T_STEPS * H_DIM);

  // 4) Persistent recurrence: 32 WGs, 256 threads, ~261 KB dynamic LDS each
  const size_t smem = (size_t)128 * 1024 * 2 + (H_DIM + 128 + 32) * sizeof(float);
  lstm_recurrence<<<NWG_REC, 256, smem, stream>>>(Ubf, pre, c0, hbuf, cnt, h_out);
}
